// SwitchMoE_39737037423281
// MI455X (gfx1250) — matrix operand order repaired and checked by reading, not compiled, not run on hardware
//
#include <hip/hip_runtime.h>
#include <hip/hip_bf16.h>

typedef __attribute__((ext_vector_type(16))) _Float16 v16h;
typedef __attribute__((ext_vector_type(8)))  _Float16 v8h;
typedef __attribute__((ext_vector_type(8)))  float    v8f;

#define D_MODEL 768
#define D_FF    3072
#define NEXP    8
#define NTOK    4096
#define GATE_BLOCKS (NTOK / 256)

// ---------------------------------------------------------------------------
// ws layout (bytes):
//   [0,   32)      : int   counts[8]
//   [64,  576)     : float partial[GATE_BLOCKS*8]
//   [1024,132096)  : int   lists[8][4096]
//   [256K, +37.7M) : f16   W1T[e][f][d]   (transposed, converted)
//   [.. ,  +37.7M) : f16   W2T[e][d][f]   (transposed, converted)
// ---------------------------------------------------------------------------
#define WS_OFF_W1T   ((size_t)1 << 18)
#define W_ELEMS      ((size_t)NEXP * D_MODEL * D_FF)
#define WS_OFF_W2T   (WS_OFF_W1T + W_ELEMS * 2)
#define WS_NEEDED    (WS_OFF_W2T + W_ELEMS * 2)

__global__ void moe_init(int* __restrict__ counts) {
    if (threadIdx.x < NEXP) counts[threadIdx.x] = 0;
}

// ---------------------------------------------------------------------------
// Gating: logits -> softmax -> top-1, expert lists, loss partials.
// ---------------------------------------------------------------------------
__global__ __launch_bounds__(256)
void moe_gate(const float* __restrict__ x, const float* __restrict__ Wg,
              const float* __restrict__ bg, int* __restrict__ counts,
              int* __restrict__ lists, float* __restrict__ partial) {
    __shared__ float wgs[D_MODEL * NEXP];
    __shared__ float red[256];
    const int tid = threadIdx.x;

    for (int i = tid; i < D_MODEL * NEXP; i += 256) wgs[i] = Wg[i];
    __syncthreads();

    const int t = blockIdx.x * 256 + tid;
    const float* xr = x + (size_t)t * D_MODEL;

    float acc[NEXP];
#pragma unroll
    for (int e = 0; e < NEXP; ++e) acc[e] = bg[e];
    for (int k = 0; k < D_MODEL; ++k) {
        const float xv = xr[k];
#pragma unroll
        for (int e = 0; e < NEXP; ++e) acc[e] += xv * wgs[k * NEXP + e];
    }

    int best = 0;
    float m = acc[0];
#pragma unroll
    for (int e = 1; e < NEXP; ++e) if (acc[e] > m) { m = acc[e]; best = e; }

    float p[NEXP];
    float s = 0.0f;
#pragma unroll
    for (int e = 0; e < NEXP; ++e) { p[e] = __expf(acc[e] - m); s += p[e]; }
    const float inv = __builtin_amdgcn_rcpf(s);
#pragma unroll
    for (int e = 0; e < NEXP; ++e) p[e] *= inv;

    const int pos = atomicAdd(&counts[best], 1);
    lists[best * NTOK + pos] = t;

    for (int e = 0; e < NEXP; ++e) {
        red[tid] = p[e];
        __syncthreads();
        for (int st = 128; st > 0; st >>= 1) {
            if (tid < st) red[tid] += red[tid + st];
            __syncthreads();
        }
        if (tid == 0) partial[blockIdx.x * NEXP + e] = red[0];
        __syncthreads();
    }
}

__global__ void moe_aux(const int* __restrict__ counts,
                        const float* __restrict__ partial,
                        float* __restrict__ loss_out) {
    if (threadIdx.x == 0 && blockIdx.x == 0) {
        float s = 0.0f;
        for (int e = 0; e < NEXP; ++e) {
            float ps = 0.0f;
            for (int b = 0; b < GATE_BLOCKS; ++b) ps += partial[b * NEXP + e];
            s += (ps / (float)NTOK) * ((float)counts[e] / (float)NTOK);
        }
        *loss_out = (float)NEXP * s;
    }
}

// ---------------------------------------------------------------------------
// Tiled transpose + f32->f16 convert:  in[R][C] f32  ->  out[C][R] f16.
// ---------------------------------------------------------------------------
__global__ __launch_bounds__(256)
void w_transpose_cvt(const float* __restrict__ in, _Float16* __restrict__ outT,
                     int R, int C) {
    __shared__ float tile[32][33];
    const float* inp = in + (size_t)blockIdx.z * R * C;
    _Float16*    op  = outT + (size_t)blockIdx.z * R * C;
    const int r0 = blockIdx.y * 32, c0 = blockIdx.x * 32;
    const int tx = threadIdx.x, ty = threadIdx.y;   // 32 x 8
#pragma unroll
    for (int i = 0; i < 32; i += 8)
        tile[ty + i][tx] = inp[(size_t)(r0 + ty + i) * C + c0 + tx];
    __syncthreads();
#pragma unroll
    for (int i = 0; i < 32; i += 8)
        op[(size_t)(c0 + ty + i) * R + r0 + tx] = (_Float16)tile[tx][ty + i];
}

// ---------------------------------------------------------------------------
// Fast FFN: M=32 token tile, pre-transposed f16 weights.
// mh = wave>>2 (constant per wave) -> one A fragment per k-step feeds
// 3 (phase-1) / 12 (phase-2) independent accumulators.  All B fragments of a
// k-step are materialized into distinct buffers BEFORE the WMMA train so the
// allocator cannot alias them and the L2 latency is exposed once per k-step.
// ---------------------------------------------------------------------------
__global__ __launch_bounds__(256)
void moe_ffn_f16w(const float* __restrict__ x,
                  const _Float16* __restrict__ W1T, const float* __restrict__ b1,
                  const _Float16* __restrict__ W2T, const float* __restrict__ b2,
                  const int* __restrict__ counts, const int* __restrict__ lists,
                  float* __restrict__ out) {
    __shared__ __align__(16) _Float16 xt[32 * D_MODEL];  // 48 KB
    __shared__ __align__(16) _Float16 hb[32 * 192];      // 12 KB
    __shared__ int toks[32];

    const int e     = blockIdx.y;
    const int ne    = counts[e];
    const int tbase = blockIdx.x * 32;
    if (tbase >= ne) return;

    const int tid  = threadIdx.x;
    const int wave = tid >> 5;
    const int lane = tid & 31;
    const int nn   = lane & 15;
    const int half = lane >> 4;
    const int mhi  = half << 3;
    const int kbA  = half * 16;
    const int kbB  = half * 16;
    const int mh   = wave >> 2;        // M half owned by this wave (0 or 1)
    const int wq   = wave & 3;         // wave index within the half
    const int arow = mh * 16 + nn;     // A-fragment source row

    if (tid < 32) {
        int idx = tbase + tid;
        if (idx >= ne) idx = tbase;
        toks[tid] = lists[e * NTOK + idx];
    }
    __syncthreads();

    for (int r = 0; r < 32; ++r) {
        const float* xr = x + (size_t)toks[r] * D_MODEL;
        for (int c = tid; c < D_MODEL; c += 256)
            xt[r * D_MODEL + c] = (_Float16)xr[c];
    }
    __syncthreads();

    const _Float16* W1e = W1T + (size_t)e * D_MODEL * D_FF;  // [f][d]
    const _Float16* W2e = W2T + (size_t)e * D_FF * D_MODEL;  // [d][f]
    const float*    b1e = b1 + (size_t)e * D_FF;
    const float*    b2e = b2 + (size_t)e * D_MODEL;

    const v8f z = {};
    v8f yacc[12];
#pragma unroll
    for (int t = 0; t < 12; ++t) yacc[t] = z;

    for (int ch = 0; ch < 16; ++ch) {
        const int fbase = ch * 192;

        // ---- phase 1: h-chunk = silu(x @ W1 + b1) ----
        {
            v8f hacc[3];
#pragma unroll
            for (int j = 0; j < 3; ++j) hacc[j] = z;
#pragma unroll
            for (int k0 = 0; k0 < D_MODEL; k0 += 32) {
                v8h lo = *(const v8h*)(xt + arow * D_MODEL + k0 + kbA);
                v8h hi = *(const v8h*)(xt + arow * D_MODEL + k0 + kbA + 8);
                v16h a;
#pragma unroll
                for (int q = 0; q < 8; ++q) { a[q] = lo[q]; a[q + 8] = hi[q]; }
                v16h bf[3];
#pragma unroll
                for (int j = 0; j < 3; ++j) {
                    const int fg = fbase + (wq * 3 + j) * 16;
                    bf[j] = *(const v16h*)(W1e + (size_t)(fg + nn) * D_MODEL + k0 + kbB);
                }
#pragma unroll
                for (int j = 0; j < 3; ++j)
                    hacc[j] = __builtin_amdgcn_wmma_f32_16x16x32_f16(
                                false, a, false, bf[j], (short)0, hacc[j], false, false);
            }
#pragma unroll
            for (int j = 0; j < 3; ++j) {
                const int fl = (wq * 3 + j) * 16;
                const float bias = b1e[fbase + fl + nn];
#pragma unroll
                for (int v = 0; v < 8; ++v) {
                    float a0 = hacc[j][v] + bias;
                    float sv = a0 * __builtin_amdgcn_rcpf(1.0f + __expf(-a0));  // SiLU
                    hb[(mh * 16 + v + mhi) * 192 + fl + nn] = (_Float16)sv;
                }
            }
        }
        __syncthreads();

        // ---- phase 2: y += h-chunk @ W2-chunk ----
#pragma unroll
        for (int k0 = 0; k0 < 192; k0 += 32) {
            v8h lo = *(const v8h*)(hb + arow * 192 + k0 + kbA);
            v8h hi = *(const v8h*)(hb + arow * 192 + k0 + kbA + 8);
            v16h a;
#pragma unroll
            for (int q = 0; q < 8; ++q) { a[q] = lo[q]; a[q + 8] = hi[q]; }
            v16h bf[12];
#pragma unroll
            for (int j = 0; j < 12; ++j) {
                const int dc = (wq * 12 + j) * 16;
                bf[j] = *(const v16h*)(W2e + (size_t)(dc + nn) * D_FF + fbase + k0 + kbB);
            }
#pragma unroll
            for (int j = 0; j < 12; ++j)
                yacc[j] = __builtin_amdgcn_wmma_f32_16x16x32_f16(
                            false, a, false, bf[j], (short)0, yacc[j], false, false);
        }
        __syncthreads();
    }

    for (int j = 0; j < 12; ++j) {
        const int dc = (wq * 12 + j) * 16;
        const float bias = b2e[dc + nn];
#pragma unroll
        for (int v = 0; v < 8; ++v) {
            const int m = mh * 16 + v + mhi;
            if (tbase + m < ne)
                out[(size_t)toks[m] * D_MODEL + dc + nn] = yacc[j][v] + bias;
        }
    }
}

// ---------------------------------------------------------------------------
// Fallback FFN (M=16, on-the-fly weight conversion) for small ws_size.
// ---------------------------------------------------------------------------
__global__ __launch_bounds__(256)
void moe_ffn(const float* __restrict__ x,
             const float* __restrict__ W1, const float* __restrict__ b1,
             const float* __restrict__ W2, const float* __restrict__ b2,
             const int* __restrict__ counts, const int* __restrict__ lists,
             float* __restrict__ out) {
    __shared__ __align__(16) _Float16 xt[16 * D_MODEL];
    __shared__ __align__(16) _Float16 hb[16 * 1024];
    __shared__ int toks[16];

    const int e     = blockIdx.y;
    const int ne    = counts[e];
    const int tbase = blockIdx.x * 16;
    if (tbase >= ne) return;

    const int tid  = threadIdx.x;
    const int wave = tid >> 5;
    const int lane = tid & 31;
    const int nn   = lane & 15;
    const int half = lane >> 4;
    const int mhi  = half << 3;
    const int kbA  = half * 16;
    const int kbB  = half * 16;

    if (tid < 16) {
        int idx = tbase + tid;
        if (idx >= ne) idx = tbase;
        toks[tid] = lists[e * NTOK + idx];
    }
    __syncthreads();

    for (int r = 0; r < 16; ++r) {
        const float* xr = x + (size_t)toks[r] * D_MODEL;
        for (int c = tid; c < D_MODEL; c += 256)
            xt[r * D_MODEL + c] = (_Float16)xr[c];
    }
    __syncthreads();

    const float* W1e = W1 + (size_t)e * D_MODEL * D_FF;
    const float* W2e = W2 + (size_t)e * D_FF * D_MODEL;
    const float* b1e = b1 + (size_t)e * D_FF;
    const float* b2e = b2 + (size_t)e * D_MODEL;

    const v8f z = {};
    v8f yacc[6];
#pragma unroll
    for (int t = 0; t < 6; ++t) yacc[t] = z;

    for (int ch = 0; ch < 3; ++ch) {
        const int fbase = ch * 1024;
        for (int t = 0; t < 8; ++t) {
            const int fl = wave * 128 + t * 16;
            const int fg = fbase + fl;
            v8f acc = z;
            for (int k0 = 0; k0 < D_MODEL; k0 += 32) {
                v8h lo = *(const v8h*)(xt + nn * D_MODEL + k0 + kbA);
                v8h hi = *(const v8h*)(xt + nn * D_MODEL + k0 + kbA + 8);
                v16h a;
#pragma unroll
                for (int q = 0; q < 8; ++q) { a[q] = lo[q]; a[q + 8] = hi[q]; }
                v16h bf;
                const float* wp = W1e + (size_t)(k0 + kbB) * D_FF + fg + nn;
#pragma unroll
                for (int q = 0; q < 16; ++q) bf[q] = (_Float16)wp[(size_t)q * D_FF];
                acc = __builtin_amdgcn_wmma_f32_16x16x32_f16(
                        false, a, false, bf, (short)0, acc, false, false);
            }
            const float bias = b1e[fg + nn];
#pragma unroll
            for (int v = 0; v < 8; ++v) {
                float a0 = acc[v] + bias;
                float sv = a0 * __builtin_amdgcn_rcpf(1.0f + __expf(-a0));
                hb[(v + mhi) * 1024 + fl + nn] = (_Float16)sv;
            }
        }
        __syncthreads();
        for (int t = 0; t < 6; ++t) {
            const int dc = wave * 96 + t * 16;
            v8f acc = yacc[t];
            for (int k0 = 0; k0 < 1024; k0 += 32) {
                v8h lo = *(const v8h*)(hb + nn * 1024 + k0 + kbA);
                v8h hi = *(const v8h*)(hb + nn * 1024 + k0 + kbA + 8);
                v16h a;
#pragma unroll
                for (int q = 0; q < 8; ++q) { a[q] = lo[q]; a[q + 8] = hi[q]; }
                v16h bf;
                const float* wp = W2e + (size_t)(fbase + k0 + kbB) * D_MODEL + dc + nn;
#pragma unroll
                for (int q = 0; q < 16; ++q) bf[q] = (_Float16)wp[(size_t)q * D_MODEL];
                acc = __builtin_amdgcn_wmma_f32_16x16x32_f16(
                        false, a, false, bf, (short)0, acc, false, false);
            }
            yacc[t] = acc;
        }
        __syncthreads();
    }

    for (int t = 0; t < 6; ++t) {
        const int dc = wave * 96 + t * 16;
        const float bias = b2e[dc + nn];
#pragma unroll
        for (int v = 0; v < 8; ++v) {
            const int m = v + mhi;
            if (tbase + m < ne)
                out[(size_t)toks[m] * D_MODEL + dc + nn] = yacc[t][v] + bias;
        }
    }
}

// ---------------------------------------------------------------------------
extern "C" void kernel_launch(void* const* d_in, const int* in_sizes, int n_in,
                              void* d_out, int out_size, void* d_ws, size_t ws_size,
                              hipStream_t stream) {
    (void)in_sizes; (void)n_in; (void)out_size;
    const float* x  = (const float*)d_in[0];
    const float* Wg = (const float*)d_in[1];
    const float* bg = (const float*)d_in[2];
    const float* W1 = (const float*)d_in[3];
    const float* b1 = (const float*)d_in[4];
    const float* W2 = (const float*)d_in[5];
    const float* b2 = (const float*)d_in[6];
    float* out = (float*)d_out;

    char*  ws      = (char*)d_ws;
    int*   counts  = (int*)ws;
    float* partial = (float*)(ws + 64);
    int*   lists   = (int*)(ws + 1024);

    moe_init<<<1, 64, 0, stream>>>(counts);
    moe_gate<<<GATE_BLOCKS, 256, 0, stream>>>(x, Wg, bg, counts, lists, partial);
    moe_aux<<<1, 32, 0, stream>>>(counts, partial, out + (size_t)NTOK * D_MODEL);

    if (ws_size >= WS_NEEDED) {
        _Float16* W1T = (_Float16*)(ws + WS_OFF_W1T);
        _Float16* W2T = (_Float16*)(ws + WS_OFF_W2T);
        // W1[e]: [768][3072] -> W1T[e]: [3072][768]
        w_transpose_cvt<<<dim3(D_FF / 32, D_MODEL / 32, NEXP), dim3(32, 8), 0, stream>>>(
            W1, W1T, D_MODEL, D_FF);
        // W2[e]: [3072][768] -> W2T[e]: [768][3072]
        w_transpose_cvt<<<dim3(D_MODEL / 32, D_FF / 32, NEXP), dim3(32, 8), 0, stream>>>(
            W2, W2T, D_FF, D_MODEL);
        dim3 grid(NTOK / 32, NEXP);
        moe_ffn_f16w<<<grid, 256, 0, stream>>>(x, W1T, b1, W2T, b2, counts, lists, out);
    } else {
        dim3 grid(NTOK / 16, NEXP);
        moe_ffn<<<grid, 256, 0, stream>>>(x, W1, b1, W2, b2, counts, lists, out);
    }
}
